// MultiHeadAttentionLayer_10187662426123
// MI455X (gfx1250) — matrix operand order repaired and checked by reading, not compiled, not run on hardware
//
#include <hip/hip_runtime.h>
#include <hip/hip_bf16.h>

// ---------------------------------------------------------------------------
// Transformer encoder layer (MHA + BN + FFN + BN) for gfx1250 (MI455X).
// GEMMs / attention use V_WMMA_F32_16X16X32_F16 (wave32); global->LDS tile
// staging uses GLOBAL_LOAD_ASYNC_TO_LDS_B128 (ASYNCcnt) with double buffering.
// ---------------------------------------------------------------------------

typedef __attribute__((ext_vector_type(16))) _Float16 v16h;
typedef __attribute__((ext_vector_type(8)))  float    v8f;

#define MROWS 32768          // B*N
#define DDIM  512            // D = FF_HID = H*DK
#define NELE  16777216       // MROWS * DDIM

__device__ __forceinline__ v8f wmma_f16(v16h a, v16h b, v8f c) {
  // D = A(16x32 f16) * B(32x16 f16) + C(16x16 f32)
  return __builtin_amdgcn_wmma_f32_16x16x32_f16(
      /*neg_a=*/false, a, /*neg_b=*/false, b,
      /*c_mod=*/(short)0, c, /*reuse_a=*/false, /*reuse_b=*/false);
}

// ds_swizzle SWAPX16 (group-of-32: and=0x1f, xor=0x10) -> exchange lanes l <-> l+16
__device__ __forceinline__ float xswap16(float x) {
  return __int_as_float(__builtin_amdgcn_ds_swizzle(__float_as_int(x), 0x401F));
}

__device__ __forceinline__ void ld_frag(v16h* f, const _Float16* lo, const _Float16* hi) {
  ((uint4*)f)[0] = *(const uint4*)lo;   // halves 0..7
  ((uint4*)f)[1] = *(const uint4*)hi;   // halves 8..15
}

// Async global -> LDS copy of 16 bytes. The low 32 bits of a generic pointer
// into LDS are the LDS byte address.
__device__ __forceinline__ void async_ld_b128(_Float16* lds_dst, const _Float16* gsrc) {
  unsigned l = (unsigned)(size_t)(void*)lds_dst;
  asm volatile("global_load_async_to_lds_b128 %0, %1, off"
               :: "v"(l), "v"(gsrc) : "memory");
}
__device__ __forceinline__ void wait_async0() {
  asm volatile("s_wait_asynccnt 0x0" ::: "memory");
}

// ------------------------------ elementwise --------------------------------

__global__ void cast_f32_f16(const float* __restrict__ src,
                             _Float16* __restrict__ dst, int n) {
  int stride = gridDim.x * blockDim.x;
  for (int i = blockIdx.x * blockDim.x + threadIdx.x; i < n; i += stride)
    dst[i] = (_Float16)src[i];
}

// Pack 6 weight matrices into f16, each 512x512, GEMM row-major B layout:
//  mat 0..2 : Wq/Wk/Wv (H,D,DK) -> B[d][h*64+k]
//  mat 3    : Wo (H,DK,D)       -> already row (h*64+k), col d : plain cast
//  mat 4    : ff_w1 (FF,D)      -> B[d][j] = w1[j][d]   (transpose)
//  mat 5    : ff_w2 (D,FF)      -> B[j][d] = w2[d][j]   (transpose)
__global__ void pack_weights(const float* __restrict__ Wq, const float* __restrict__ Wk,
                             const float* __restrict__ Wv, const float* __restrict__ Wo,
                             const float* __restrict__ W1, const float* __restrict__ W2,
                             _Float16* __restrict__ dst) {
  int idx = blockIdx.x * blockDim.x + threadIdx.x;
  if (idx >= 6 * 262144) return;
  int mat = idx >> 18;
  int r   = idx & 262143;
  int row = r >> 9;        // 0..511
  int col = r & 511;       // 0..511
  float v;
  if (mat <= 2) {
    const float* W = (mat == 0) ? Wq : (mat == 1) ? Wk : Wv;
    int h = col >> 6, k = col & 63;
    v = W[h * 32768 + row * 64 + k];
  } else if (mat == 3) {
    v = Wo[r];
  } else if (mat == 4) {
    v = W1[col * 512 + row];
  } else {
    v = W2[col * 512 + row];
  }
  dst[idx] = (_Float16)v;
}

// ------------------------------ WMMA GEMM ----------------------------------
// C[M x 512] = A[M x 512] * B[512 x 512], A/B f16 row-major.
// Block tile 128x128, K-step 32, 8 waves (4x2), wave tile 32x64 -> 8 WMMA accs.
// Double-buffered async global->LDS staging.

__global__ __launch_bounds__(256) void gemm_wmma(
    const _Float16* __restrict__ A, const _Float16* __restrict__ Bm,
    float* __restrict__ outF, _Float16* __restrict__ outH,
    const float* __restrict__ bias, int relu, int headmajor) {
  __shared__ _Float16 sA[2][128 * 32];   // 2 x 8 KB
  __shared__ _Float16 sB[2][32 * 128];   // 2 x 8 KB

  const int m0 = blockIdx.x * 128;
  const int n0 = blockIdx.y * 128;
  const int tid  = threadIdx.x;
  const int wave = tid >> 5, lane = tid & 31;
  const int wm = wave >> 1, wn = wave & 1;        // 4x2 wave grid
  const int l = lane & 15, hi = lane >> 4;

  v8f acc[2][4];
  v8f z = {};
  for (int ms = 0; ms < 2; ++ms)
    for (int ns = 0; ns < 4; ++ns) acc[ms][ns] = z;

  auto stage = [&](int buf, int k0) {
    // A tile: 128 rows x 32 halves = 512 chunks of 8 halves (16B) -> 2/thread
    for (int c = tid; c < 512; c += 256) {
      int row = c >> 2, sub = c & 3;
      async_ld_b128(&sA[buf][row * 32 + sub * 8],
                    &A[(size_t)(m0 + row) * 512 + k0 + sub * 8]);
    }
    // B tile: 32 rows x 128 halves = 512 chunks
    for (int c = tid; c < 512; c += 256) {
      int row = c >> 4, sub = c & 15;
      async_ld_b128(&sB[buf][row * 128 + sub * 8],
                    &Bm[(size_t)(k0 + row) * 512 + n0 + sub * 8]);
    }
  };

  stage(0, 0);

  for (int k0 = 0; k0 < 512; k0 += 32) {
    const int cur = (k0 >> 5) & 1;
    wait_async0();        // this thread's copies into buf[cur] are done
    __syncthreads();      // everyone's copies done; prior reads of buf[cur^1] done
    if (k0 + 32 < 512) stage(cur ^ 1, k0 + 32);  // overlap with compute

    v16h af[2];
    for (int ms = 0; ms < 2; ++ms) {
      const _Float16* pa = &sA[cur][(wm * 32 + ms * 16 + l) * 32 + 8 * hi];
      ld_frag(&af[ms], pa, pa + 16);  // K = 8hi+0..7 , 16+8hi+0..7
    }
    for (int ns = 0; ns < 4; ++ns) {
      const _Float16* pb = &sB[cur][wn * 64 + ns * 16 + l];
      v16h bf;
      for (int s = 0; s < 16; ++s) bf[s] = pb[((s & 8) * 2 + 8 * hi + (s & 7)) * 128];  // K = 8hi+0..7 , 16+8hi+0..7
      for (int ms = 0; ms < 2; ++ms)
        acc[ms][ns] = wmma_f16(af[ms], bf, acc[ms][ns]);
    }
  }

  // store: D[m][n], lane(l,hi): n = l, vgpr j: m = j + 8*hi
  for (int ms = 0; ms < 2; ++ms) {
    for (int ns = 0; ns < 4; ++ns) {
      for (int j = 0; j < 8; ++j) {
        int m = m0 + wm * 32 + ms * 16 + 8 * hi + j;
        int n = n0 + wn * 64 + ns * 16 + l;
        float v = acc[ms][ns][j];
        if (bias) v += bias[n];
        if (relu) v = fmaxf(v, 0.0f);
        if (outF) outF[(size_t)m * 512 + n] = v;
        if (outH) {
          if (headmajor) {
            // Q/K/V: [h][m][k]  (h = n>>6, k = n&63)
            outH[(size_t)(n >> 6) * ((size_t)MROWS * 64) + (size_t)m * 64 + (n & 63)] =
                (_Float16)v;
          } else {
            outH[(size_t)m * 512 + n] = (_Float16)v;
          }
        }
      }
    }
  }
}

// ------------------------------ attention ----------------------------------
// grid = H*B*16 blocks, 64 threads (2 waves); each wave owns 16 query rows.
// Q/K/V layout: [h][b*512+n][64] f16.

__global__ __launch_bounds__(64) void attn_wmma(
    const _Float16* __restrict__ Q, const _Float16* __restrict__ Km,
    const _Float16* __restrict__ V, _Float16* __restrict__ heads) {
  __shared__ _Float16 sc[2][16 * 512];  // per-wave score rows (f16), 32 KB
  __shared__ _Float16 kv[2048];         // K-tile (transposed) / V-tile, 4 KB
  __shared__ float    rsum[2][16];

  const int bid = blockIdx.x;
  const int qb = bid & 15;
  const int hb = bid >> 4;
  const int b = hb & 63, h = hb >> 6;
  const int tid = threadIdx.x, wave = tid >> 5, lane = tid & 31;
  const int l = lane & 15, hi = lane >> 4;

  const size_t HM = (size_t)MROWS * 64;
  const _Float16* Qb = Q + (size_t)h * HM + (size_t)b * 512 * 64;
  const _Float16* Kb = Km + (size_t)h * HM + (size_t)b * 512 * 64;
  const _Float16* Vb = V + (size_t)h * HM + (size_t)b * 512 * 64;
  const int q0 = qb * 32 + wave * 16;
  _Float16* mysc = sc[wave];

  // Q fragments (rows q0..q0+15, DK=64 -> 2 K-steps), kept in registers
  v16h qa[2];
  {
    const _Float16* qrow = Qb + (size_t)(q0 + l) * 64;
    for (int ds = 0; ds < 2; ++ds) {
      const _Float16* p = qrow + ds * 32 + 8 * hi;
      ld_frag(&qa[ds], p, p + 16);
    }
  }

  // Phase A: scores = (Q K^T) * 0.125, tile-by-tile over 16 key blocks of 32
  for (int kb = 0; kb < 16; ++kb) {
    __syncthreads();
    // K tile transposed into LDS: kv[d*32 + kk] = K[kb*32+kk][d]
    for (int i = tid; i < 2048; i += 64) {
      int kk = i & 31, d = i >> 5;
      kv[d * 32 + kk] = Kb[(size_t)(kb * 32 + kk) * 64 + d];
    }
    __syncthreads();
    for (int ns = 0; ns < 2; ++ns) {
      v8f a = {};
      for (int ds = 0; ds < 2; ++ds) {
        const _Float16* pb = &kv[ds * 32 * 32 + ns * 16 + l];
        v16h bf;
        for (int s = 0; s < 16; ++s) bf[s] = pb[((s & 8) * 2 + 8 * hi + (s & 7)) * 32];
        a = wmma_f16(qa[ds], bf, a);
      }
      for (int j = 0; j < 8; ++j)
        mysc[(j + 8 * hi) * 512 + kb * 32 + ns * 16 + l] =
            (_Float16)(a[j] * 0.125f);
    }
  }

  // Phase B: softmax over 512 keys per row (lane pair l / l+16 splits a row)
  {
    int row = lane & 15;
    _Float16* pr = &mysc[row * 512 + hi * 256];
    float mx = -3.0e38f;
    for (int c = 0; c < 256; ++c) mx = fmaxf(mx, (float)pr[c]);
    mx = fmaxf(mx, xswap16(mx));
    float s = 0.0f;
    for (int c = 0; c < 256; ++c) {
      float e = __expf((float)pr[c] - mx);
      s += e;
      pr[c] = (_Float16)e;
    }
    s += xswap16(s);
    rsum[wave][row] = s;
  }

  // Phase C: O = P * V  (contraction over 512 keys, 16 blocks of 32)
  v8f oa[4];
  {
    v8f z = {};
    for (int ns = 0; ns < 4; ++ns) oa[ns] = z;
  }
  for (int kb = 0; kb < 16; ++kb) {
    __syncthreads();
    // V tile staged async: kv[kk*64 + d]
    for (int c = tid; c < 256; c += 64) {
      int kk = c >> 3, sub = c & 7;
      async_ld_b128(&kv[kk * 64 + sub * 8],
                    &Vb[(size_t)(kb * 32 + kk) * 64 + sub * 8]);
    }
    if (kb + 1 < 16)  // warm next tile (global_prefetch_b8)
      __builtin_prefetch(&Vb[(size_t)((kb + 1) * 32 + (tid >> 1)) * 64], 0, 0);
    wait_async0();
    __syncthreads();
    v16h pa;
    const _Float16* pp = &mysc[l * 512 + kb * 32 + 8 * hi];
    ld_frag(&pa, pp, pp + 16);
    for (int ns = 0; ns < 4; ++ns) {
      const _Float16* pb = &kv[ns * 16 + l];
      v16h bf;
      for (int s = 0; s < 16; ++s) bf[s] = pb[((s & 8) * 2 + 8 * hi + (s & 7)) * 64];
      oa[ns] = wmma_f16(pa, bf, oa[ns]);
    }
  }

  // store heads: [b*512+q][h*64+d], normalized by row sum
  for (int ns = 0; ns < 4; ++ns) {
    for (int j = 0; j < 8; ++j) {
      int m = q0 + 8 * hi + j;
      int d = ns * 16 + l;
      float v = oa[ns][j] / rsum[wave][8 * hi + j];
      heads[(size_t)(b * 512 + m) * 512 + h * 64 + d] = (_Float16)v;
    }
  }
}

// ------------------------------ batch norm ---------------------------------
// Deterministic two-stage stats. val = 2*(p + q)   (q optional).

__global__ __launch_bounds__(256) void bn_stats_part(
    const float* __restrict__ p, const float* __restrict__ q,
    float* __restrict__ partial) {
  const int blk = blockIdx.x;          // 256 blocks, 128 rows each
  const int tid = threadIdx.x;
  const int r0 = blk * 128;
  for (int d = tid; d < 512; d += 256) {
    float s = 0.0f, s2 = 0.0f;
    for (int r = 0; r < 128; ++r) {
      size_t e = (size_t)(r0 + r) * 512 + d;
      float v = p[e];
      if (q) v += q[e];
      v *= 2.0f;
      s += v;
      s2 += v * v;
    }
    partial[(size_t)blk * 1024 + d] = s;
    partial[(size_t)blk * 1024 + 512 + d] = s2;
  }
}

__global__ void bn_stats_reduce(const float* __restrict__ partial,
                                float* __restrict__ stats) {
  int d = blockIdx.x * blockDim.x + threadIdx.x;
  if (d < 1024) {
    float s = 0.0f;
    for (int b = 0; b < 256; ++b) s += partial[(size_t)b * 1024 + d];
    stats[d] = s;
  }
}

__global__ void bn_apply(const float* __restrict__ p, const float* __restrict__ q,
                         const float* __restrict__ stats,
                         const float* __restrict__ w, const float* __restrict__ bb,
                         float* __restrict__ outF, _Float16* __restrict__ outH) {
  const float invM = 1.0f / (float)MROWS;
  int stride = gridDim.x * blockDim.x;
  for (int e = blockIdx.x * blockDim.x + threadIdx.x; e < NELE; e += stride) {
    int d = e & 511;
    float m = stats[d] * invM;
    float var = stats[512 + d] * invM - m * m;
    float v = p[e];
    if (q) v += q[e];
    v *= 2.0f;
    float r = (v - m) * rsqrtf(var + 1e-5f) * w[d] + bb[d];
    if (outF) outF[e] = r;
    if (outH) outH[e] = (_Float16)r;
  }
}

// ------------------------------ launcher -----------------------------------

extern "C" void kernel_launch(void* const* d_in, const int* in_sizes, int n_in,
                              void* d_out, int out_size, void* d_ws, size_t ws_size,
                              hipStream_t stream) {
  const float* x   = (const float*)d_in[0];
  const float* Wq  = (const float*)d_in[1];
  const float* Wk  = (const float*)d_in[2];
  const float* Wv  = (const float*)d_in[3];
  const float* Wo  = (const float*)d_in[4];
  const float* n1w = (const float*)d_in[5];
  const float* n1b = (const float*)d_in[6];
  const float* n2w = (const float*)d_in[7];
  const float* n2b = (const float*)d_in[8];
  const float* fw1 = (const float*)d_in[9];
  const float* fb1 = (const float*)d_in[10];
  const float* fw2 = (const float*)d_in[11];
  const float* fb2 = (const float*)d_in[12];

  // workspace layout
  _Float16* xh    = (_Float16*)d_ws;            // 16.7M halves
  _Float16* Wpk   = xh + (size_t)NELE;          // 6 * 262144 halves
  _Float16* Qh    = Wpk + 6 * 262144;           // [h][m][64]
  _Float16* Kh    = Qh + (size_t)NELE;
  _Float16* Vh    = Kh + (size_t)NELE;
  _Float16* heads = Vh + (size_t)NELE;
  _Float16* hh    = heads + (size_t)NELE;       // h (post-BN1) in f16
  _Float16* ffh   = hh + (size_t)NELE;          // FFN hidden f16
  float* t0    = (float*)(ffh + (size_t)NELE);  // attn-out / ff2-out f32
  float* t1    = t0 + (size_t)NELE;             // h (post-BN1) f32
  float* stats = t1 + (size_t)NELE;             // 1024 floats
  float* part  = stats + 1024;                  // 256*1024 floats

  // 1) cast x, pack weights
  cast_f32_f16<<<4096, 256, 0, stream>>>(x, xh, NELE);
  pack_weights<<<6144, 256, 0, stream>>>(Wq, Wk, Wv, Wo, fw1, fw2, Wpk);

  dim3 gg(MROWS / 128, 512 / 128);  // 256 x 4

  // 2) Q/K/V projections (head-major f16 outputs)
  gemm_wmma<<<gg, 256, 0, stream>>>(xh, Wpk + 0 * 262144, nullptr, Qh, nullptr, 0, 1);
  gemm_wmma<<<gg, 256, 0, stream>>>(xh, Wpk + 1 * 262144, nullptr, Kh, nullptr, 0, 1);
  gemm_wmma<<<gg, 256, 0, stream>>>(xh, Wpk + 2 * 262144, nullptr, Vh, nullptr, 0, 1);

  // 3) attention
  attn_wmma<<<8 * 64 * 16, 64, 0, stream>>>(Qh, Kh, Vh, heads);

  // 4) output projection -> t0 (f32)
  gemm_wmma<<<gg, 256, 0, stream>>>(heads, Wpk + 3 * 262144, t0, nullptr, nullptr, 0, 0);

  // 5) BN1 on 2*t0 -> t1 (f32) and hh (f16)
  bn_stats_part<<<256, 256, 0, stream>>>(t0, nullptr, part);
  bn_stats_reduce<<<4, 256, 0, stream>>>(part, stats);
  bn_apply<<<4096, 256, 0, stream>>>(t0, nullptr, stats, n1w, n1b, t1, hh);

  // 6) FFN: relu(h @ w1^T + b1) @ w2^T + b2 -> t0 (f32)
  gemm_wmma<<<gg, 256, 0, stream>>>(hh, Wpk + 4 * 262144, nullptr, ffh, fb1, 1, 0);
  gemm_wmma<<<gg, 256, 0, stream>>>(ffh, Wpk + 5 * 262144, t0, nullptr, fb2, 0, 0);

  // 7) BN2 on 2*(t1 + t0) -> d_out
  bn_stats_part<<<256, 256, 0, stream>>>(t1, t0, part);
  bn_stats_reduce<<<4, 256, 0, stream>>>(part, stats);
  bn_apply<<<4096, 256, 0, stream>>>(t1, t0, stats, n2w, n2b, (float*)d_out, nullptr);
}
